// FactorizedSpectralConv_56573309223202
// MI455X (gfx1250) — matrix operand order repaired and checked by reading, not compiled, not run on hardware
//
#include <hip/hip_runtime.h>
#include <hip/hip_bf16.h>

typedef __attribute__((ext_vector_type(16))) __bf16 v16bf;
typedef __attribute__((ext_vector_type(8)))  float  v8f;

#define N_LEN 8192
#define BATCH 32
#define CIN   64
#define COUT  64
#define FMODES 64
#define RANK  16
#define MROWS (BATCH*CIN)      // 2048
#define KSPLIT 8
#define INV_SQRT_N 0.011048543456039806f   // 1/sqrt(8192)

// ---------------- workspace layout (bytes) ----------------
#define WS_XPART 0x000000   // KSPLIT*2048*128 f32 = 8 MB
#define WS_YBF   0x800000   // 2048*128 bf16       = 512 KB
#define WS_WR    0x880000   // 64*64 f32           = 16 KB
#define WS_WI    0x884000   // 64*64 f32           = 16 KB
#define WS_BFWD  0x888000   // 128 x 8192 bf16     = 2 MB   (forward basis, B^T)
#define WS_GINV  0xA88000   // 8192 x 128 bf16     = 2 MB   (inverse basis, B^T)

// Async copy global -> LDS, tracked by ASYNCcnt (CDNA5 §08 async-tensor path).
// The LDS *pointer* is converted to its 32-bit addrspace(3) offset inside the
// helper so the shared object escapes into the asm -> the "memory" clobber
// aliases it and the compiler cannot fold subsequent LDS reads to undef.
__device__ __forceinline__ void async_g2l_b128(void* lds_dst, const void* gaddr) {
  asm volatile("global_load_async_to_lds_b128 %0, %1, off"
               :: "v"((unsigned)(uintptr_t)lds_dst), "v"(gaddr)
               : "memory");
}
__device__ __forceinline__ void async_g2l_b128x2(void* lds_dst, const void* gaddr) {
  asm volatile("global_load_async_to_lds_b128 %0, %1, off\n\t"
               "global_load_async_to_lds_b128 %0, %1, off offset:16"
               :: "v"((unsigned)(uintptr_t)lds_dst), "v"(gaddr)
               : "memory");
}
__device__ __forceinline__ void wait_asynccnt0() {
  asm volatile("s_wait_asynccnt 0x0" ::: "memory");
}

// ---- kernel: w = (U*S)V complex, 64x64 ----
__global__ __launch_bounds__(256) void k_weights(const float* __restrict__ Ur,
                                                 const float* __restrict__ Ui,
                                                 const float* __restrict__ Vr,
                                                 const float* __restrict__ Vi,
                                                 const float* __restrict__ S,
                                                 float* __restrict__ wr,
                                                 float* __restrict__ wi) {
  int id = blockIdx.x * 256 + threadIdx.x;      // 4096 total
  int i = id >> 6, o = id & 63;
  float ar = 0.f, ai = 0.f;
  for (int r = 0; r < RANK; ++r) {
    float s  = S[r];
    float ur = Ur[i * RANK + r] * s, ui = Ui[i * RANK + r] * s;
    float vr = Vr[r * COUT + o],     vi = Vi[r * COUT + o];
    ar += ur * vr - ui * vi;
    ai += ur * vi + ui * vr;
  }
  wr[id] = ar;
  wi[id] = ai;
}

// ---- kernel: build bf16 DFT bases (forward transposed + inverse transposed) ----
__global__ __launch_bounds__(256) void k_basis(__bf16* __restrict__ bfwd,
                                               __bf16* __restrict__ ginv) {
  int id = blockIdx.x * 256 + threadIdx.x;      // FMODES*N_LEN threads
  int k = id >> 13;
  int n = id & (N_LEN - 1);
  float ang = (6.28318530717958647692f / (float)N_LEN) * (float)(k * n);
  float s, c;
  __sincosf(ang, &s, &c);
  // forward (rfft, ortho): Xr = sum x*cos/sqrtN ; Xi = -sum x*sin/sqrtN
  bfwd[(size_t)k * N_LEN + n]            = (__bf16)(c * INV_SQRT_N);
  bfwd[(size_t)(FMODES + k) * N_LEN + n] = (__bf16)(-s * INV_SQRT_N);
  // inverse (irfft, ortho, truncated): out = sum_k ck*(Yr*cos - Yi*sin)/sqrtN
  float ck = (k == 0) ? 1.0f : 2.0f;
  ginv[(size_t)n * 128 + k]          = (__bf16)(ck * c * INV_SQRT_N);
  ginv[(size_t)n * 128 + FMODES + k] = (__bf16)((k == 0) ? 0.0f : (-2.0f * s * INV_SQRT_N));
}

// ---- kernel: forward GEMM  Xpart[ks] = x(2048x8192) * Bfwd(8192x128) ----
// grid: 16 row-blocks * KSPLIT, 256 threads (8 waves, 16 rows each)
// Double-buffered LDS B-chunk staged via async global->LDS.
__global__ __launch_bounds__(256) void k_fwd(const float* __restrict__ x,
                                             const __bf16* __restrict__ bfwd,
                                             float* __restrict__ xpart) {
  __shared__ uint4 ldsB[2][512];                    // 2 x 8 KB: [col 0..127][32 K bf16]
  const int rb   = blockIdx.x >> 3;                 // 0..15
  const int ks   = blockIdx.x & 7;                  // 0..7
  const int wave = threadIdx.x >> 5;
  const int lane = threadIdx.x & 31;
  const int row0 = rb * 128 + wave * 16;
  const int m    = lane & 15;
  const int kb   = (lane < 16) ? 0 : 8;             // A-frag K base (ISA 7.12.2)
  const int bkb  = (lane < 16) ? 0 : 16;            // B-frag byte base within col chunk
  const float* arow = x + (size_t)(row0 + m) * N_LEN;

  const v8f vzero = {0.f, 0.f, 0.f, 0.f, 0.f, 0.f, 0.f, 0.f};
  v8f acc[8];
#pragma unroll
  for (int t = 0; t < 8; ++t) acc[t] = vzero;

  const int stage_c = threadIdx.x >> 1;             // staging col 0..127
  const int stage_h = threadIdx.x & 1;              // 32-byte half of the 64-byte col chunk
  const int kc0 = ks * (N_LEN / KSPLIT);
  const unsigned stage_off = (unsigned)(stage_c * 64 + stage_h * 32);

  // prologue: stage chunk 0 into buffer 0
  async_g2l_b128x2((char*)&ldsB[0][0] + stage_off,
                   (const char*)(bfwd + (size_t)stage_c * N_LEN + kc0) + stage_h * 32);
  wait_asynccnt0();
  __syncthreads();

  for (int it = 0; it < 32; ++it) {                 // 1024 K per block, 32 per chunk
    const int kc  = kc0 + it * 32;
    const int cur = it & 1;

    // stage next chunk into the other buffer (async, in flight during compute)
    if (it + 1 < 32) {
      async_g2l_b128x2((char*)&ldsB[(it + 1) & 1][0] + stage_off,
                       (const char*)(bfwd + (size_t)stage_c * N_LEN + kc + 32) + stage_h * 32);
    }

    // A fragment: rows of x, convert f32 -> bf16 on the fly
    union { v16bf v; __bf16 e[16]; } a;
    {
      const float4* ap = (const float4*)(arow + kc + kb);
      float4 f0 = ap[0], f1 = ap[1], f2 = ap[4], f3 = ap[5];
      a.e[0]=(__bf16)f0.x; a.e[1]=(__bf16)f0.y; a.e[2]=(__bf16)f0.z; a.e[3]=(__bf16)f0.w;
      a.e[4]=(__bf16)f1.x; a.e[5]=(__bf16)f1.y; a.e[6]=(__bf16)f1.z; a.e[7]=(__bf16)f1.w;
      a.e[8]=(__bf16)f2.x; a.e[9]=(__bf16)f2.y; a.e[10]=(__bf16)f2.z; a.e[11]=(__bf16)f2.w;
      a.e[12]=(__bf16)f3.x; a.e[13]=(__bf16)f3.y; a.e[14]=(__bf16)f3.z; a.e[15]=(__bf16)f3.w;
    }

    // preload all 8 B fragments, then 8 back-to-back WMMAs (waits pipeline down)
    union BF { v16bf v; uint4 q[2]; };
    BF bf[8];
    const char* lbase = (const char*)&ldsB[cur][0] + m * 64 + bkb;
#pragma unroll
    for (int t = 0; t < 8; ++t) {
      bf[t].q[0] = *(const uint4*)(lbase + t * 1024);
      bf[t].q[1] = *(const uint4*)(lbase + t * 1024 + 32);
    }
#pragma unroll
    for (int t = 0; t < 8; ++t)
      acc[t] = __builtin_amdgcn_wmma_f32_16x16x32_bf16(
          false, a.v, false, bf[t].v, (short)0, acc[t], false, false);

    wait_asynccnt0();
    __syncthreads();
  }

  // deterministic epilogue: plain stores into this K-split's partial buffer
  float* xp = xpart + (size_t)ks * MROWS * 128;
  const int rbase = row0 + ((lane < 16) ? 0 : 8);
#pragma unroll
  for (int t = 0; t < 8; ++t) {
    const int c = t * 16 + m;
#pragma unroll
    for (int v = 0; v < 8; ++v)
      xp[(size_t)(rbase + v) * 128 + c] = acc[t][v];
  }
}

// ---- kernel: reduce K-split partials + complex channel mix, emit Y in bf16 ----
// grid: 32 blocks (one per batch b), 256 threads
__global__ __launch_bounds__(256) void k_mix(const float* __restrict__ xpart,
                                             const float* __restrict__ wr,
                                             const float* __restrict__ wi,
                                             __bf16* __restrict__ ybf) {
  __shared__ float4 sXv[CIN * 128 / 4];             // 32 KB  [i][k | 64+k]
  __shared__ float4 sWrv[CIN * COUT / 4];           // 16 KB
  __shared__ float4 sWiv[CIN * COUT / 4];           // 16 KB
  float* sX  = (float*)sXv;
  float* sWr = (float*)sWrv;
  float* sWi = (float*)sWiv;
  const int b = blockIdx.x;
  const int t = threadIdx.x;

  for (int j = t; j < CIN * 128 / 4; j += 256) {    // sum the KSPLIT partials
    float4 a = {0.f, 0.f, 0.f, 0.f};
#pragma unroll
    for (int s = 0; s < KSPLIT; ++s) {
      const float4 p = ((const float4*)(xpart + ((size_t)s * MROWS + (size_t)b * CIN) * 128))[j];
      a.x += p.x; a.y += p.y; a.z += p.z; a.w += p.w;
    }
    sXv[j] = a;
  }
  for (int j = t; j < CIN * COUT / 4; j += 256) {
    sWrv[j] = ((const float4*)wr)[j];
    sWiv[j] = ((const float4*)wi)[j];
  }
  __syncthreads();

  for (int p = t * 16; p < t * 16 + 16; ++p) {      // 4096 (o,k) pairs total
    const int o = p >> 6, k = p & 63;
    float yr = 0.f, yi = 0.f;
    for (int i = 0; i < CIN; ++i) {
      const float xr = sX[i * 128 + k], xi = sX[i * 128 + 64 + k];
      const float wre = sWr[i * COUT + o], wim = sWi[i * COUT + o];
      yr += wre * xr - wim * xi;
      yi += wre * xi + wim * xr;
    }
    ybf[((size_t)b * COUT + o) * 128 + k]      = (__bf16)yr;
    ybf[((size_t)b * COUT + o) * 128 + 64 + k] = (__bf16)yi;
  }
}

// ---- kernel: inverse GEMM  out(2048x8192) = Y(2048x128) * Ginv(128x8192) ----
// grid: 16 row-blocks * 8 col-strips, 256 threads (8 waves share col strip)
// Double-buffered async-staged B tiles; A resident in VGPRs for the whole kernel.
__global__ __launch_bounds__(256) void k_inv(const __bf16* __restrict__ ybf,
                                             const __bf16* __restrict__ ginv,
                                             float* __restrict__ out) {
  __shared__ uint4 ldsB[2][256];                    // 2 x 4 KB: [col 0..15][128 K bf16]
  const int rblk = blockIdx.x >> 3;                 // 0..15
  const int cs   = blockIdx.x & 7;                  // 0..7
  const int wave = threadIdx.x >> 5;
  const int lane = threadIdx.x & 31;
  const int row0 = rblk * 128 + wave * 16;
  const int m    = lane & 15;
  const int kb   = (lane < 16) ? 0 : 8;             // A-frag K base
  const int bkb  = (lane < 16) ? 0 : 16;            // B-frag byte base

  // A fragments (16 rows x 128 K of Y, bf16) resident in VGPRs
  union { v16bf v; uint4 q[2]; } afr[4];
  const __bf16* yrow = ybf + (size_t)(row0 + m) * 128;
#pragma unroll
  for (int q = 0; q < 4; ++q) {
    afr[q].q[0] = *(const uint4*)(yrow + q * 32 + kb);
    afr[q].q[1] = *(const uint4*)(yrow + q * 32 + kb + 16);
  }

  const int sc = threadIdx.x >> 4;                  // staging col 0..15
  const int sp = threadIdx.x & 15;                  // 16-byte piece of the 256-byte col
  const int nbase = cs * 1024;
  const unsigned stage_off = (unsigned)(sc * 256 + sp * 16);
  const v8f vzero = {0.f, 0.f, 0.f, 0.f, 0.f, 0.f, 0.f, 0.f};

  // prologue: stage tile 0 into buffer 0
  async_g2l_b128((char*)&ldsB[0][0] + stage_off,
                 (const char*)(ginv + (size_t)(nbase + sc) * 128) + sp * 16);
  wait_asynccnt0();
  __syncthreads();

  for (int ct = 0; ct < 64; ++ct) {                 // 64 col tiles per strip
    const int n0  = nbase + ct * 16;
    const int cur = ct & 1;

    if (ct + 1 < 64) {
      async_g2l_b128((char*)&ldsB[(ct + 1) & 1][0] + stage_off,
                     (const char*)(ginv + (size_t)(n0 + 16 + sc) * 128) + sp * 16);
    }

    v8f acc = vzero;
#pragma unroll
    for (int q = 0; q < 4; ++q) {
      union { v16bf v; uint4 u[2]; } bfr;
      const char* base = (const char*)&ldsB[cur][0] + m * 256 + q * 64 + bkb;
      bfr.u[0] = *(const uint4*)(base);
      bfr.u[1] = *(const uint4*)(base + 32);
      acc = __builtin_amdgcn_wmma_f32_16x16x32_bf16(
          false, afr[q].v, false, bfr.v, (short)0, acc, false, false);
    }

    const int rb2 = row0 + ((lane < 16) ? 0 : 8);
    const int n   = n0 + m;
#pragma unroll
    for (int v = 0; v < 8; ++v)
      out[(size_t)(rb2 + v) * N_LEN + n] = acc[v];

    wait_asynccnt0();
    __syncthreads();
  }
}

extern "C" void kernel_launch(void* const* d_in, const int* in_sizes, int n_in,
                              void* d_out, int out_size, void* d_ws, size_t ws_size,
                              hipStream_t stream) {
  const float* x  = (const float*)d_in[0];
  const float* Ur = (const float*)d_in[1];
  const float* Ui = (const float*)d_in[2];
  const float* Vr = (const float*)d_in[3];
  const float* Vi = (const float*)d_in[4];
  const float* S  = (const float*)d_in[5];

  char* ws = (char*)d_ws;
  float*  xpart = (float*)(ws + WS_XPART);
  __bf16* ybf   = (__bf16*)(ws + WS_YBF);
  float*  wr    = (float*)(ws + WS_WR);
  float*  wi    = (float*)(ws + WS_WI);
  __bf16* bfwd  = (__bf16*)(ws + WS_BFWD);
  __bf16* ginv  = (__bf16*)(ws + WS_GINV);
  float*  out   = (float*)d_out;

  k_weights<<<(CIN * COUT) / 256, 256, 0, stream>>>(Ur, Ui, Vr, Vi, S, wr, wi);
  k_basis<<<(FMODES * N_LEN) / 256, 256, 0, stream>>>(bfwd, ginv);
  k_fwd<<<16 * KSPLIT, 256, 0, stream>>>(x, bfwd, xpart);
  k_mix<<<BATCH, 256, 0, stream>>>(xpart, wr, wi, ybf);
  k_inv<<<16 * 8, 256, 0, stream>>>(ybf, ginv, out);
}
